// Router_10746008175522
// MI455X (gfx1250) — matrix operand order repaired and checked by reading, not compiled, not run on hardware
//
#include <hip/hip_runtime.h>
#include <hip/hip_bf16.h>

typedef __attribute__((ext_vector_type(16))) _Float16 v16h;
typedef __attribute__((ext_vector_type(8)))  _Float16 v8h;
typedef __attribute__((ext_vector_type(8)))  float    v8f;
typedef __attribute__((ext_vector_type(4)))  float    v4f;

#define DIM_IN       256
#define HIDDEN       32
#define NUM_EXPERTS  64
#define TEMP_INV     10.0f
#define WAVES_PER_BLOCK   8
#define TILES_PER_WAVE    8     // 8 x 16 = 128 tokens per wave
#define TOKENS_PER_BLOCK  (WAVES_PER_BLOCK * TILES_PER_WAVE * 16)   // 1024

// butterfly reductions over the 16-lane half-wave (xor masks 1,2,4,8 never cross bit 4)
__device__ __forceinline__ float hmax16(float v) {
#pragma unroll
    for (int m = 1; m <= 8; m <<= 1) v = fmaxf(v, __shfl_xor(v, m, 32));
    return v;
}
__device__ __forceinline__ float hsum16(float v) {
#pragma unroll
    for (int m = 1; m <= 8; m <<= 1) v += __shfl_xor(v, m, 32);
    return v;
}
__device__ __forceinline__ void hargmax16(float& v, int& i) {
#pragma unroll
    for (int m = 1; m <= 8; m <<= 1) {
        float ov = __shfl_xor(v, m, 32);
        int   oi = __shfl_xor(i, m, 32);
        if (ov > v || (ov == v && oi < i)) { v = ov; i = oi; }
    }
}

__device__ __forceinline__ float fast_tanh(float y) {
    // tanh(y) = 1 - 2/(exp(2y)+1); saturates correctly for |y| large
    float t = __expf(2.0f * y);
    return 1.0f - 2.0f / (t + 1.0f);
}

__global__ __launch_bounds__(256)
void router_wmma_kernel(const float* __restrict__ x,
                        const float* __restrict__ W1,
                        const float* __restrict__ b1,
                        const float* __restrict__ W2,
                        const float* __restrict__ b2,
                        float* __restrict__ out,
                        int n_tokens)
{
    // W1 as f16 in WMMA B layout: [chunk(8)][ntile(2)][lane(32)][e(16)]
    __shared__ __attribute__((aligned(64))) _Float16 ldsW1[8 * 2 * 32 * 16];  // 16 KB
    // W2 as f16 in WMMA B layout: [tile(4)][lane(32)][e(16)]
    __shared__ __attribute__((aligned(64))) _Float16 ldsW2[4 * 32 * 16];      // 4 KB
    // per-wave h scratch, row-major [16 tokens][32 hidden] f16
    __shared__ __attribute__((aligned(64))) _Float16 ldsH[WAVES_PER_BLOCK][16 * 32]; // 8 KB

    const int tid = threadIdx.x;

    // ---- stage W1 into LDS, converted to f16, pre-swizzled to B layout ----
    // B layout (16x16x32 f16, 32K x 16N): lane<16 holds K={0..7,16..23}, lane>=16 K={8..15,24..31}; col N = lane&15
#pragma unroll
    for (int f = tid; f < 8 * 2 * 32 * 16; f += 256) {
        int e    = f & 15;
        int lane = (f >> 4) & 31;
        int nt   = (f >> 9) & 1;
        int c    = f >> 10;
        int krow = 32 * c + ((e & 8) ? 16 : 0) + ((lane >= 16) ? 8 : 0) + (e & 7);
        int col  = nt * 16 + (lane & 15);
        ldsW1[f] = (_Float16)W1[krow * HIDDEN + col];
    }
#pragma unroll
    for (int f = tid; f < 4 * 32 * 16; f += 256) {
        int e    = f & 15;
        int lane = (f >> 4) & 31;
        int t    = f >> 9;
        int krow = ((e & 8) ? 16 : 0) + ((lane >= 16) ? 8 : 0) + (e & 7);
        int col  = t * 16 + (lane & 15);
        ldsW2[f] = (_Float16)W2[krow * NUM_EXPERTS + col];
    }
    __syncthreads();

    const int wave   = tid >> 5;
    const int lane   = tid & 31;
    const int laneN  = lane & 15;     // column / row-in-halfwave id
    const int laneHi = lane >> 4;     // 0: lanes 0-15, 1: lanes 16-31

    // loop-invariant operands hoisted out of the tile loop
    const float bias0 = b1[laneN];
    const float bias1 = b1[16 + laneN];
    float bb2[4];
#pragma unroll
    for (int t = 0; t < 4; ++t) bb2[t] = b2[t * 16 + laneN];
    v16h w2frag[4];
#pragma unroll
    for (int t = 0; t < 4; ++t)
        w2frag[t] = *(const v16h*)(ldsW2 + (t * 32 + lane) * 16);

    const long waveBase = (long)blockIdx.x * TOKENS_PER_BLOCK + wave * (TILES_PER_WAVE * 16);
    _Float16* __restrict__ hbuf = &ldsH[wave][0];

#pragma clang loop unroll(disable)
    for (int tile = 0; tile < TILES_PER_WAVE; ++tile) {
        const long tokBase = waveBase + tile * 16;
        long myRow = tokBase + laneN;
        if (myRow >= n_tokens) myRow = n_tokens - 1;          // clamp tail loads
        const float* __restrict__ xrow = x + myRow * DIM_IN;

        // warm L2/WGP$ for the next tile's row (speculative, clamped)
        {
            long pfRow = myRow + 16;
            if (pfRow >= n_tokens) pfRow = n_tokens - 1;
            __builtin_prefetch(x + pfRow * DIM_IN, 0, 3);
        }

        // ---- GEMM1: h(16x32) = x(16x256) * W1(256x32), 8 K-chunks of 32 ----
        v8f c0 = {}; v8f c1 = {};
#pragma unroll
        for (int c = 0; c < 8; ++c) {
            // A 16-bit layout: lane<16 -> K = 32c + {0..7, 16..23}; lane>=16 -> +8
            const int off = c * 32 + laneHi * 8;
            v4f a0 = *(const v4f*)(xrow + off);
            v4f a1 = *(const v4f*)(xrow + off + 4);
            v4f a2 = *(const v4f*)(xrow + off + 16);
            v4f a3 = *(const v4f*)(xrow + off + 20);
            v16h a;
#pragma unroll
            for (int e = 0; e < 4; ++e) {
                a[e]      = (_Float16)a0[e];
                a[4 + e]  = (_Float16)a1[e];
                a[8 + e]  = (_Float16)a2[e];
                a[12 + e] = (_Float16)a3[e];
            }
            const v16h b0  = *(const v16h*)(ldsW1 + ((c * 2 + 0) * 32 + lane) * 16);
            const v16h b1v = *(const v16h*)(ldsW1 + ((c * 2 + 1) * 32 + lane) * 16);
            c0 = __builtin_amdgcn_wmma_f32_16x16x32_f16(false, a, false, b0,  (short)0, c0, false, false);
            c1 = __builtin_amdgcn_wmma_f32_16x16x32_f16(false, a, false, b1v, (short)0, c1, false, false);
        }

        // ---- bias + tanh, write h (f16) to wave-private LDS (C-layout -> row-major) ----
#pragma unroll
        for (int r = 0; r < 8; ++r) {
            const int M = r + laneHi * 8;
            hbuf[M * 32 + laneN]      = (_Float16)fast_tanh(c0[r] + bias0);
            hbuf[M * 32 + 16 + laneN] = (_Float16)fast_tanh(c1[r] + bias1);
        }
        // same-wave producer/consumer through LDS: drain DS counter (no barrier needed)
        asm volatile("s_wait_dscnt 0x0" ::: "memory");

        // ---- read h back in A layout: row M=laneN, K = {off..off+7, 16+off..16+off+7} ----
        const int hoff = laneHi * 8;
        const _Float16* hr = hbuf + laneN * 32;
        v8h lo = *(const v8h*)(hr + hoff);
        v8h hi = *(const v8h*)(hr + 16 + hoff);
        v16h ah;
#pragma unroll
        for (int e = 0; e < 8; ++e) { ah[e] = lo[e]; ah[8 + e] = hi[e]; }

        // ---- GEMM2: logits(16x64) = h(16x32) * W2(32x64), 4 N-tiles ----
        v8f acc[4];
#pragma unroll
        for (int t = 0; t < 4; ++t) {
            acc[t] = v8f{};
            acc[t] = __builtin_amdgcn_wmma_f32_16x16x32_f16(false, ah, false, w2frag[t],
                                                            (short)0, acc[t], false, false);
        }

        // ---- per-row softmax / top-2 / renormalize ----
#pragma unroll
        for (int r = 0; r < 8; ++r) {
            float v[4]; int idx[4];
#pragma unroll
            for (int t = 0; t < 4; ++t) {
                v[t]   = (acc[t][r] + bb2[t]) * TEMP_INV;
                idx[t] = t * 16 + laneN;
            }
            float m = fmaxf(fmaxf(v[0], v[1]), fmaxf(v[2], v[3]));
            m = hmax16(m);

            float e[4];
            float s = 0.0f;
#pragma unroll
            for (int t = 0; t < 4; ++t) { e[t] = __expf(v[t] - m); s += e[t]; }
            s = hsum16(s);

            // top-1 (lowest index wins ties)
            float bv = v[0]; int bi = idx[0];
#pragma unroll
            for (int t = 1; t < 4; ++t)
                if (v[t] > bv) { bv = v[t]; bi = idx[t]; }
            hargmax16(bv, bi);

            // top-2: exclude bi locally, reduce again
            float sv = -3.0e38f; int si = 0x7fffffff;
#pragma unroll
            for (int t = 0; t < 4; ++t)
                if (idx[t] != bi && (v[t] > sv || (v[t] == sv && idx[t] < si))) { sv = v[t]; si = idx[t]; }
            hargmax16(sv, si);

            // w[e] = exp(l-m) / (e1 + e2 + 1e-9*S) on kept experts, else 0
            const float denom = __expf(bv - m) + __expf(sv - m) + 1e-9f * s;
            const float rden  = 1.0f / denom;

            const long row = tokBase + r + laneHi * 8;
            if (row < n_tokens) {
                float* __restrict__ orow = out + row * NUM_EXPERTS + laneN;
#pragma unroll
                for (int t = 0; t < 4; ++t) {
                    const bool keep = (idx[t] == bi) | (idx[t] == si);
                    orow[t * 16] = keep ? e[t] * rden : 0.0f;
                }
            }
        }
    }
}

extern "C" void kernel_launch(void* const* d_in, const int* in_sizes, int n_in,
                              void* d_out, int out_size, void* d_ws, size_t ws_size,
                              hipStream_t stream) {
    const float* x  = (const float*)d_in[0];
    const float* W1 = (const float*)d_in[1];
    const float* b1 = (const float*)d_in[2];
    const float* W2 = (const float*)d_in[3];
    const float* b2 = (const float*)d_in[4];
    float* out = (float*)d_out;

    const int n_tokens = in_sizes[0] / DIM_IN;
    const int blocks = (n_tokens + TOKENS_PER_BLOCK - 1) / TOKENS_PER_BLOCK;
    router_wmma_kernel<<<blocks, 256, 0, stream>>>(x, W1, b1, W2, b2, out, n_tokens);
}
